// SwitchTransformersLayerFF_69965017252082
// MI455X (gfx1250) — matrix operand order repaired and checked by reading, not compiled, not run on hardware
//
#include <hip/hip_runtime.h>
#include <hip/hip_bf16.h>

#define D_MODEL 512
#define D_FF    2048
#define NEXP    8
#define NTOK    4096   // B*S = 2*2048
#define LN_EPS  1e-6f

typedef __attribute__((ext_vector_type(16))) __bf16          v16bf;
typedef __attribute__((ext_vector_type(8)))  float           v8f;
typedef __attribute__((ext_vector_type(8)))  unsigned short  v8u16;

__device__ __forceinline__ unsigned short f32_to_bf16(float f) {
    union { float f; unsigned u; } v; v.f = f;
    unsigned r = v.u + 0x7FFFu + ((v.u >> 16) & 1u);   // round-to-nearest-even
    return (unsigned short)(r >> 16);
}

__device__ __forceinline__ unsigned pack2_bf16(float lo, float hi) {
    return (unsigned)f32_to_bf16(lo) | ((unsigned)f32_to_bf16(hi) << 16);
}

__device__ __forceinline__ v16bf pack_bf16(v8u16 lo, v8u16 hi) {
    union { struct { v8u16 a, b; } u; v16bf v; } x;
    x.u.a = lo; x.u.b = hi;
    return x.v;
}

// ---------------------------------------------------------------- init ------
__global__ void init_counts_k(int* counts) {
    if (threadIdx.x < NEXP) counts[threadIdx.x] = 0;
}

// -------------------------------------------------- weight f32 -> bf16 ------
// One-shot per launch: stream-convert a weight tensor (8 floats / thread).
__global__ __launch_bounds__(256)
void cvt_bf16_k(const float* __restrict__ src, unsigned* __restrict__ dst) {
    const size_t i = (size_t)blockIdx.x * 256 + threadIdx.x;   // uint4 index
    const float4* s4 = (const float4*)(src + i * 8);
    const float4 a = s4[0], b = s4[1];
    uint4 o;
    o.x = pack2_bf16(a.x, a.y);
    o.y = pack2_bf16(a.z, a.w);
    o.z = pack2_bf16(b.x, b.y);
    o.w = pack2_bf16(b.z, b.w);
    ((uint4*)dst)[i] = o;
}

// ------------------------------------------------------------- router -------
// One wave32 per token: RMSNorm + bf16 normed + router logits/softmax/argmax
// + per-expert token list build (atomic; order-independent result).
__global__ __launch_bounds__(256)
void router_k(const float* __restrict__ x, const float* __restrict__ lnw,
              const float* __restrict__ wr, unsigned short* __restrict__ normed,
              float* __restrict__ prob, int* __restrict__ counts,
              int* __restrict__ tlist, float* __restrict__ out_logits,
              int* __restrict__ out_idx) {
    const int lane = threadIdx.x & 31;
    const int wv   = threadIdx.x >> 5;
    const int t    = blockIdx.x * 8 + wv;
    if (t >= NTOK) return;

    const float* xp = x + (size_t)t * D_MODEL;

    float v[16];
    float ss = 0.f;
#pragma unroll
    for (int i = 0; i < 16; ++i) {
        const int d = lane + i * 32;
        v[i] = xp[d];
        ss += v[i] * v[i];
    }
#pragma unroll
    for (int m = 16; m; m >>= 1) ss += __shfl_xor(ss, m, 32);
    const float rstd = rsqrtf(ss * (1.0f / (float)D_MODEL) + LN_EPS);

    float lg[8] = {0.f, 0.f, 0.f, 0.f, 0.f, 0.f, 0.f, 0.f};
#pragma unroll
    for (int i = 0; i < 16; ++i) {
        const int d = lane + i * 32;
        const float nv = v[i] * rstd * lnw[d];
        normed[(size_t)t * D_MODEL + d] = f32_to_bf16(nv);
        const float4* wp = (const float4*)(wr + (size_t)d * NEXP);
        const float4 w0 = wp[0], w1 = wp[1];
        lg[0] += nv * w0.x; lg[1] += nv * w0.y;
        lg[2] += nv * w0.z; lg[3] += nv * w0.w;
        lg[4] += nv * w1.x; lg[5] += nv * w1.y;
        lg[6] += nv * w1.z; lg[7] += nv * w1.w;
    }
#pragma unroll
    for (int e = 0; e < 8; ++e)
#pragma unroll
        for (int m = 16; m; m >>= 1) lg[e] += __shfl_xor(lg[e], m, 32);

    if (lane == 0) {
        float mx = lg[0]; int ai = 0;
#pragma unroll
        for (int e = 1; e < 8; ++e)
            if (lg[e] > mx) { mx = lg[e]; ai = e; }
        float s = 0.f;
#pragma unroll
        for (int e = 0; e < 8; ++e) s += __expf(lg[e] - mx);
        prob[t]    = 1.0f / s;        // softmax prob of the max logit
        out_idx[t] = ai;
#pragma unroll
        for (int e = 0; e < 8; ++e) out_logits[(size_t)t * NEXP + e] = lg[e];
        const int pos = atomicAdd(&counts[ai], 1);
        tlist[ai * NTOK + pos] = t;
    }
}

// ------------------------------------------------------- fused MoE GEMM -----
// Tile: 128 tokens x 128 cols, K-step 32, bf16 WMMA, double-buffered LDS.
// A tile: per-lane gather via GLOBAL_LOAD_ASYNC_TO_LDS_B128 (ASYNCcnt).
// B tile: pre-converted bf16 weights, register transpose into LDS.
// FUSE_OUT=false: store relu(acc) as bf16 into hbuf.
// FUSE_OUT=true : out = x + prob * acc  (f32 scatter by token id).
template <int KDIM, int LDA, int LDB, bool FUSE_OUT>
__global__ __launch_bounds__(256)
void moe_gemm_k(const unsigned short* __restrict__ Abuf,
                const unsigned short* __restrict__ Wbf,
                const int* __restrict__ counts, const int* __restrict__ tlist,
                unsigned short* __restrict__ hbuf,
                const float* __restrict__ prob, const float* __restrict__ x,
                float* __restrict__ out) {
    __shared__ __align__(16) unsigned short As[2][128 * 32];
    __shared__ __align__(16) unsigned short Bs[2][128 * 32];
    __shared__ int toks[128];

    const int e     = blockIdx.z;
    const int cnt   = counts[e];
    const int tbase = blockIdx.y * 128;
    if (tbase >= cnt) return;
    const int c0 = blockIdx.x * 128;

    const int tid  = threadIdx.x;
    const int lane = tid & 31;
    const int wv   = tid >> 5;

    if (tid < 128)
        toks[tid] = (tbase + tid < cnt) ? tlist[e * NTOK + tbase + tid] : -1;
    __syncthreads();

    const unsigned short* We = Wbf + (size_t)e * KDIM * LDB;

    // loader coords
    const int ar = tid >> 1;             // A: row 0..127 (2 threads/row)
    const int ak = (tid & 1) * 16;       // A: 16 halfs each
    const int bk = tid >> 3;             // B: k 0..31 (8 threads/k)
    const int bn = (tid & 7) * 16;       // B: 16 cols each

    // clamp padding rows to token 0: garbage computed, never stored
    const int tok_a = toks[ar] < 0 ? 0 : toks[ar];
    const unsigned short* aga = Abuf + (size_t)tok_a * LDA + ak;

    const v8f zf = {0.f, 0.f, 0.f, 0.f, 0.f, 0.f, 0.f, 0.f};
    v8f acc[8];
#pragma unroll
    for (int n = 0; n < 8; ++n) acc[n] = zf;

    auto load_tile = [&](int kk, int s) {
        // ---- A: two async b128 global->LDS (per-lane gather, ASYNCcnt)
        const unsigned short* ga = aga + kk;
        const unsigned lds = (unsigned)(unsigned long long)&As[s][ar * 32 + ak];
        asm volatile("global_load_async_to_lds_b128 %0, %1, off"
                     :: "v"(lds), "v"(ga) : "memory");
        asm volatile("global_load_async_to_lds_b128 %0, %1, off offset:16"
                     :: "v"(lds), "v"(ga) : "memory");
        // ---- B: bf16 weights row-major -> transposed LDS [n][k]
        const unsigned short* bp = We + (size_t)(kk + bk) * LDB + c0 + bn;
        const v8u16 b0 = ((const v8u16*)bp)[0];
        const v8u16 b1 = ((const v8u16*)bp)[1];
#pragma unroll
        for (int j = 0; j < 8; ++j) {
            Bs[s][(bn + j) * 32 + bk]     = b0[j];
            Bs[s][(bn + 8 + j) * 32 + bk] = b1[j];
        }
    };

    constexpr int NITER = KDIM / 32;
    load_tile(0, 0);

    for (int i = 0; i < NITER; ++i) {
        asm volatile("s_wait_asynccnt 0x0" ::: "memory");
        __syncthreads();
        if (i + 1 < NITER) load_tile((i + 1) * 32, (i + 1) & 1);

        const int s    = i & 1;
        const int arow = 16 * wv + (lane & 15);
        const int ka   = (lane >> 4) * 8;
        const v16bf afrag =
            pack_bf16(*(const v8u16*)&As[s][arow * 32 + ka],
                      *(const v8u16*)&As[s][arow * 32 + ka + 16]);
        const int kb = (lane >> 4) * 8;
        const int cb = lane & 15;
#pragma unroll
        for (int n = 0; n < 8; ++n) {
            const int c = n * 16 + cb;
            const v16bf bfrag =
                pack_bf16(*(const v8u16*)&Bs[s][c * 32 + kb],
                          *(const v8u16*)&Bs[s][c * 32 + kb + 16]);
            acc[n] = __builtin_amdgcn_wmma_f32_16x16x32_bf16(
                false, afrag, false, bfrag, (short)0, acc[n], false, false);
        }
    }

    // ---- epilogue (C layout: lane holds col N=lane%16; VGPR r -> M=r+8*(lane/16))
    const int cb    = lane & 15;
    const int rbase = 16 * wv + 8 * (lane >> 4);
#pragma unroll
    for (int n = 0; n < 8; ++n)
#pragma unroll
        for (int r = 0; r < 8; ++r) {
            const int tok = toks[rbase + r];
            if (tok >= 0) {
                if (FUSE_OUT) {
                    const size_t o = (size_t)tok * LDB + c0 + n * 16 + cb;
                    out[o] = x[o] + prob[tok] * acc[n][r];
                } else {
                    float vv = acc[n][r];
                    vv = vv > 0.f ? vv : 0.f;
                    hbuf[(size_t)tok * LDB + c0 + n * 16 + cb] = f32_to_bf16(vv);
                }
            }
        }
}

// ------------------------------------------------------------ launch --------
extern "C" void kernel_launch(void* const* d_in, const int* in_sizes, int n_in,
                              void* d_out, int out_size, void* d_ws, size_t ws_size,
                              hipStream_t stream) {
    (void)in_sizes; (void)n_in; (void)out_size; (void)ws_size;

    const float* x   = (const float*)d_in[0];   // [2,2048,512]
    const float* lnw = (const float*)d_in[1];   // [512]
    const float* wr  = (const float*)d_in[2];   // [512,8]
    const float* wi  = (const float*)d_in[3];   // [8,512,2048]
    const float* wo  = (const float*)d_in[4];   // [8,2048,512]
    // d_in[5] = output_router_logits flag (always emit all outputs)

    float* out        = (float*)d_out;                                 // [N,512]
    float* out_logits = out + (size_t)NTOK * D_MODEL;                  // [N,8]
    int*   out_idx    = (int*)(out + (size_t)NTOK * D_MODEL +
                               (size_t)NTOK * NEXP);                   // [N]

    // workspace carve (all offsets 256B-aligned)
    char* w = (char*)d_ws;
    int*            counts = (int*)w;                                    // 32 B
    float*          prob   = (float*)(w + 256);                          // 16 KB
    int*            tlist  = (int*)(w + 16640);                          // 128 KB
    unsigned short* normed = (unsigned short*)(w + 147712);              // 4 MB
    unsigned short* hbuf   = (unsigned short*)(w + 4342016);             // 16 MB
    unsigned short* wibf   = (unsigned short*)(w + 21119232);            // 16.78 MB
    unsigned short* wobf   = (unsigned short*)(w + 37896448);            // 16.78 MB

    constexpr int WELEM = NEXP * D_MODEL * D_FF;                         // 8.39M

    init_counts_k<<<1, 32, 0, stream>>>(counts);
    cvt_bf16_k<<<WELEM / (256 * 8), 256, 0, stream>>>(wi, (unsigned*)wibf);
    cvt_bf16_k<<<WELEM / (256 * 8), 256, 0, stream>>>(wo, (unsigned*)wobf);
    router_k<<<NTOK / 8, 256, 0, stream>>>(x, lnw, wr, normed, prob, counts,
                                           tlist, out_logits, out_idx);
    // h = relu(normed @ wi[e])
    moe_gemm_k<D_MODEL, D_MODEL, D_FF, false>
        <<<dim3(D_FF / 128, NTOK / 128, NEXP), 256, 0, stream>>>(
            normed, wibf, counts, tlist, hbuf, nullptr, nullptr, nullptr);
    // out = x + prob * (h @ wo[e])
    moe_gemm_k<D_FF, D_FF, D_MODEL, true>
        <<<dim3(D_MODEL / 128, NTOK / 128, NEXP), 256, 0, stream>>>(
            hbuf, wobf, counts, tlist, nullptr, prob, x, out);
}
